// MaxPoolAggregator_6854767804436
// MI455X (gfx1250) — matrix operand order repaired and checked by reading, not compiled, not run on hardware
//
#include <hip/hip_runtime.h>
#include <hip/hip_bf16.h>
#include <stdint.h>

#define N_NODES 50000
#define N_EDGES 800000
#define D_IN    128
#define D_HID   512
#define D_OUT   128
#define D_CAT   (D_IN + D_HID)

#define EDGES_PER_BLOCK 64   // 4 M-tiles of 16; amortizes B-fragment reads 4x

typedef __attribute__((ext_vector_type(16))) _Float16 v16h;
typedef __attribute__((ext_vector_type(8)))  _Float16 v8h;
typedef __attribute__((ext_vector_type(8)))  float    v8f;

static __device__ __forceinline__ v16h cat8(v8h lo, v8h hi) {
  return __builtin_shufflevector(lo, hi, 0,1,2,3,4,5,6,7,8,9,10,11,12,13,14,15);
}

// ---------------------------------------------------------------------------
// Pack fc_w [128,512] (row-major, K x N) into per-lane WMMA B fragments:
// bpack[kchunk(4)][ntile(32)][lane(32)][elem(16)], lane<16 -> K=kc*32+0..15,
// lane>=16 -> K=kc*32+16..31, column N = ntile*16 + lane%16.
// ---------------------------------------------------------------------------
__global__ void k_pack_fc(const float* __restrict__ fcw, _Float16* __restrict__ bpack) {
  const int idx = blockIdx.x * 256 + threadIdx.x;
  if (idx >= 4 * 32 * 32 * 16) return;
  const int i    = idx & 15;
  const int lane = (idx >> 4) & 31;
  const int nt   = (idx >> 9) & 31;
  const int kc   = idx >> 14;
  const int K = kc * 32 + ((lane < 16) ? 0 : 16) + i;
  const int N = nt * 16 + (lane & 15);
  bpack[idx] = (_Float16)fcw[K * D_HID + N];
}

// Pack weights_matrix [640,128] likewise: wpack[kchunk(20)][ntile(8)][lane(32)][16]
__global__ void k_pack_w(const float* __restrict__ w, _Float16* __restrict__ wpack) {
  const int idx = blockIdx.x * 256 + threadIdx.x;
  if (idx >= 20 * 8 * 32 * 16) return;
  const int i    = idx & 15;
  const int lane = (idx >> 4) & 31;
  const int nt   = (idx >> 9) & 7;
  const int kc   = idx >> 12;
  const int K = kc * 32 + ((lane < 16) ? 0 : 16) + i;
  const int N = nt * 16 + (lane & 15);
  wpack[idx] = (_Float16)w[K * D_OUT + N];
}

__global__ void k_init_agg(float4* __restrict__ agg) {
  const size_t i = (size_t)blockIdx.x * blockDim.x + threadIdx.x;
  if (i < (size_t)N_NODES * D_HID / 4) agg[i] = make_float4(0.f, 0.f, 0.f, 0.f);
}

// ---------------------------------------------------------------------------
// Fused edge MLP + scatter-max.
// Block = 256 threads (8 waves), handles 64 edges (4 M-tiles of 16).
// Each wave owns 4 N-tiles (64 hidden cols); per K-chunk it loads its 4 B
// fragments once and reuses them against 4 A fragments -> 16 WMMAs/chunk,
// 64 v_wmma_f32_16x16x32_f16 per wave total.
// Epilogue: +bias, ReLU, unsigned atomic-max scatter into agg (L2-resident;
// ReLU outputs >= 0 so u32 bit ordering == float ordering, and the zero init
// reproduces segment_max's empty-segment zero fill + max(agg,0)).
// ---------------------------------------------------------------------------
__global__ void __launch_bounds__(256) k_edge_mlp(
    const float* __restrict__ x, const _Float16* __restrict__ bpack,
    const float* __restrict__ bias, const long long* __restrict__ esrc,
    const long long* __restrict__ etrg, unsigned int* __restrict__ agg)
{
  __shared__ __align__(16) _Float16 As[EDGES_PER_BLOCK * D_IN];  // 16 KB
  __shared__ int srcs[EDGES_PER_BLOCK];
  const int tid = threadIdx.x;
  const int e0  = blockIdx.x * EDGES_PER_BLOCK;

  if (tid < EDGES_PER_BLOCK) srcs[tid] = (int)esrc[e0 + tid];
  {
    // Stage gathered A tile (64 edges x 128 feats) in LDS as f16.
    const int r0 = tid >> 4;             // 0..15
    const int c0 = (tid & 15) * 8;       // 0..120
    #pragma unroll
    for (int rr = 0; rr < 4; ++rr) {
      const int row = r0 + rr * 16;
      const long long trg = etrg[e0 + row];
      const float* xr = x + (size_t)trg * D_IN + c0;
      #pragma unroll
      for (int i = 0; i < 8; ++i) As[row * D_IN + c0 + i] = (_Float16)xr[i];
    }
  }
  __syncthreads();

  const int wave = tid >> 5;
  const int lane = tid & 31;
  const int arow = lane & 15;
  const int kh   = (lane < 16) ? 0 : 16;  // A-fragment K sub-offset: half h holds k = 16h + 0..15, as bpack

  v8f acc[4][4] = {};                      // [mtile][ntile] : 128 VGPRs
  #pragma unroll
  for (int k = 0; k < 4; ++k) {
    const _Float16* bb = bpack + (size_t)(((k * 32) + wave * 4) * 32 + lane) * 16;
    const v16h b0 = *(const v16h*)(bb + 0 * 32 * 16);
    const v16h b1 = *(const v16h*)(bb + 1 * 32 * 16);
    const v16h b2 = *(const v16h*)(bb + 2 * 32 * 16);
    const v16h b3 = *(const v16h*)(bb + 3 * 32 * 16);
    #pragma unroll
    for (int mt = 0; mt < 4; ++mt) {
      const _Float16* ab = &As[(mt * 16 + arow) * D_IN + k * 32 + kh];
      const v16h a = cat8(*(const v8h*)ab, *(const v8h*)(ab + 8));
      acc[mt][0] = __builtin_amdgcn_wmma_f32_16x16x32_f16(false, a, false, b0, (short)0, acc[mt][0], false, false);
      acc[mt][1] = __builtin_amdgcn_wmma_f32_16x16x32_f16(false, a, false, b1, (short)0, acc[mt][1], false, false);
      acc[mt][2] = __builtin_amdgcn_wmma_f32_16x16x32_f16(false, a, false, b2, (short)0, acc[mt][2], false, false);
      acc[mt][3] = __builtin_amdgcn_wmma_f32_16x16x32_f16(false, a, false, b3, (short)0, acc[mt][3], false, false);
    }
  }

  // C/D layout: VGPR r -> (M = r + (lane<16?0:8), N = ntile*16 + lane%16)
  const int moff = (lane < 16) ? 0 : 8;
  const int ncol = lane & 15;
  #pragma unroll
  for (int nt = 0; nt < 4; ++nt) {
    const int h = (wave * 4 + nt) * 16 + ncol;
    const float bv = bias[h];
    #pragma unroll
    for (int mt = 0; mt < 4; ++mt) {
      #pragma unroll
      for (int r = 0; r < 8; ++r) {
        float v = acc[mt][nt][r] + bv;
        v = v > 0.f ? v : 0.f;                       // ReLU >= 0
        atomicMax(agg + (size_t)srcs[mt * 16 + r + moff] * D_HID + h,
                  __float_as_uint(v));
      }
    }
  }
}

// ---------------------------------------------------------------------------
// out = [x | agg] @ W   -> [50000, 128]
// Block = 256 threads, 16 node rows; 8 waves x 1 N-tile; K = 640 -> 20 chunks.
// ---------------------------------------------------------------------------
__global__ void __launch_bounds__(256) k_out_gemm(
    const float* __restrict__ x, const float* __restrict__ agg,
    const _Float16* __restrict__ wpack, float* __restrict__ out)
{
  __shared__ __align__(16) _Float16 As[16 * D_CAT];   // 20 KB
  const int tid = threadIdx.x;
  const int m0  = blockIdx.x * 16;

  for (int idx = tid; idx < 16 * D_CAT; idx += 256) {
    const int r = idx / D_CAT;
    const int c = idx - r * D_CAT;
    const float v = (c < D_IN) ? x[(size_t)(m0 + r) * D_IN + c]
                               : agg[(size_t)(m0 + r) * D_HID + (c - D_IN)];
    As[idx] = (_Float16)v;
  }
  __syncthreads();

  const int wave = tid >> 5;
  const int lane = tid & 31;
  const int arow = lane & 15;
  const int kh   = (lane < 16) ? 0 : 16;

  v8f acc = {};
  #pragma unroll
  for (int k = 0; k < 20; ++k) {
    const _Float16* ab = &As[arow * D_CAT + k * 32 + kh];
    const v16h a = cat8(*(const v8h*)ab, *(const v8h*)(ab + 8));
    const v16h b = *(const v16h*)(wpack + (size_t)((k * 8 + wave) * 32 + lane) * 16);
    acc = __builtin_amdgcn_wmma_f32_16x16x32_f16(false, a, false, b, (short)0, acc, false, false);
  }

  const int n  = wave * 16 + (lane & 15);
  const int mb = m0 + ((lane < 16) ? 0 : 8);
  #pragma unroll
  for (int r = 0; r < 8; ++r)
    out[(size_t)(mb + r) * D_OUT + n] = acc[r];
}

// ---------------------------------------------------------------------------
extern "C" void kernel_launch(void* const* d_in, const int* in_sizes, int n_in,
                              void* d_out, int out_size, void* d_ws, size_t ws_size,
                              hipStream_t stream) {
  const float*     x    = (const float*)d_in[0];                 // [50000,128]
  const float*     fcw  = (const float*)d_in[1];                 // [128,512]
  const float*     fcb  = (const float*)d_in[2];                 // [512]
  const float*     w    = (const float*)d_in[3];                 // [640,128]
  const long long* adj  = (const long long*)d_in[4];             // [2,800000] int64
  float*           out  = (float*)d_out;                         // [50000,128]

  char* ws = (char*)d_ws;
  float*    agg   = (float*)ws;                                  // 102,400,000 B
  _Float16* bpack = (_Float16*)(ws + (size_t)N_NODES * D_HID * 4);          // 131,072 B
  _Float16* wpack = (_Float16*)(ws + (size_t)N_NODES * D_HID * 4 + 131072); // 163,840 B

  k_pack_fc<<<(4 * 32 * 32 * 16 + 255) / 256, 256, 0, stream>>>(fcw, bpack);
  k_pack_w <<<(20 * 8 * 32 * 16 + 255) / 256, 256, 0, stream>>>(w, wpack);
  k_init_agg<<<((N_NODES * D_HID / 4) + 255) / 256, 256, 0, stream>>>((float4*)agg);

  k_edge_mlp<<<N_EDGES / EDGES_PER_BLOCK, 256, 0, stream>>>(x, bpack, fcb, adj,
                                                            adj + N_EDGES,
                                                            (unsigned int*)agg);
  k_out_gemm<<<N_NODES / 16, 256, 0, stream>>>(x, agg, wpack, out);
}
